// PolylineSubgraphNetwork_46162308497569
// MI455X (gfx1250) — compile-verified, matrix operand order checked by reading, not run on hardware
//
#include <hip/hip_runtime.h>
#include <hip/hip_bf16.h>

#define NROWS 500000
#define NP    50000
#define INDIM 9
#define HD    128
#define LNEPS 1e-5f
#define WT_STRIDE 264   // 256 K halves + 8 pad -> lane stride = 4 LDS banks
#define WT_ELEMS  (HD * WT_STRIDE)          // 33792 halves = 67584 bytes

typedef __attribute__((ext_vector_type(16))) _Float16 v16h;
typedef __attribute__((ext_vector_type(8)))  float    v8f;

struct alignas(16) H8 { _Float16 h[8]; };

union AFrag { v16h v; _Float16 e[16]; H8 c2[2]; };

#if __has_builtin(__builtin_amdgcn_sched_group_barrier)
#define SCHED_DS_READS(n)  __builtin_amdgcn_sched_group_barrier(0x100, (n), 0)
#define SCHED_WMMAS(n)     __builtin_amdgcn_sched_group_barrier(0x008, (n), 0)
#else
#define SCHED_DS_READS(n)
#define SCHED_WMMAS(n)
#endif

// ---------------------------------------------------------------------------
// init: zero seg buffers + output features, emit unique_polylines tail
// ---------------------------------------------------------------------------
__global__ __launch_bounds__(256) void vn_init(float* __restrict__ seg0,
                                               float* __restrict__ seg1,
                                               float* __restrict__ feat,
                                               float* __restrict__ tail,
                                               int tail_mode) {
  int i = blockIdx.x * blockDim.x + threadIdx.x;
  if (i < NP * HD) {
    seg0[i] = 0.0f;
    seg1[i] = 0.0f;
    feat[i] = 0.0f;
  }
  if (i < NP) {
    if (tail_mode == 2)      ((long long*)tail)[i] = (long long)i;  // int64 flat
    else if (tail_mode == 1) tail[i] = (float)i;                    // f32 flat
  }
}

// ---------------------------------------------------------------------------
// prep: convert W1/W2 (f32 [256][128] row-major) into f16 transposed
// [col][k] layout with padded stride, resident in workspace (L2-hot, 66 KB ea)
// ---------------------------------------------------------------------------
__global__ __launch_bounds__(256) void vn_prep(const float* __restrict__ W1,
                                               const float* __restrict__ W2,
                                               _Float16* __restrict__ wt1,
                                               _Float16* __restrict__ wt2) {
  int i = blockIdx.x * blockDim.x + threadIdx.x;
  if (i < 2 * HD * HD) {
    const int k = i >> 7, col = i & (HD - 1);
    wt1[col * WT_STRIDE + k] = (_Float16)W1[i];
    wt2[col * WT_STRIDE + k] = (_Float16)W2[i];
  }
}

// ---------------------------------------------------------------------------
// layer0: [N,9] @ [9,128] + bias -> LayerNorm -> ReLU ; one wave per row
// ---------------------------------------------------------------------------
__global__ __launch_bounds__(256) void vn_layer0(const float* __restrict__ x,
                                                 const int*   __restrict__ ids,
                                                 const float* __restrict__ W0,
                                                 const float* __restrict__ b0,
                                                 const float* __restrict__ g0,
                                                 const float* __restrict__ be0,
                                                 _Float16*    __restrict__ hout,
                                                 float*       __restrict__ seg0) {
  __shared__ float w[INDIM * HD];
  __shared__ float bb[HD], gg[HD], bbe[HD];
  for (int i = threadIdx.x; i < INDIM * HD; i += blockDim.x) w[i] = W0[i];
  for (int i = threadIdx.x; i < HD; i += blockDim.x) {
    bb[i] = b0[i]; gg[i] = g0[i]; bbe[i] = be0[i];
  }
  __syncthreads();

  const int wave = threadIdx.x >> 5;
  const int lane = threadIdx.x & 31;
  const int row  = blockIdx.x * 8 + wave;
  if (row >= NROWS) return;

  float xr[INDIM];
#pragma unroll
  for (int k = 0; k < INDIM; ++k) xr[k] = x[(size_t)row * INDIM + k];

  float acc[4];
#pragma unroll
  for (int j = 0; j < 4; ++j) {
    const int col = lane + 32 * j;
    float a = bb[col];
#pragma unroll
    for (int k = 0; k < INDIM; ++k) a = fmaf(xr[k], w[k * HD + col], a);
    acc[j] = a;
  }

  // wave-wide LayerNorm over 128 values (4 per lane, 32 lanes)
  float s = 0.0f, q = 0.0f;
#pragma unroll
  for (int j = 0; j < 4; ++j) { s += acc[j]; q += acc[j] * acc[j]; }
#pragma unroll
  for (int m = 1; m < 32; m <<= 1) {
    s += __shfl_xor(s, m, 32);
    q += __shfl_xor(q, m, 32);
  }
  const float mu  = s * (1.0f / HD);
  const float var = q * (1.0f / HD) - mu * mu;
  const float rs  = rsqrtf(var + LNEPS);

  const int id = ids[row];
#pragma unroll
  for (int j = 0; j < 4; ++j) {
    const int col = lane + 32 * j;
    float v = (acc[j] - mu) * rs * gg[col] + bbe[col];
    v = fmaxf(v, 0.0f);                                // ReLU -> v >= 0
    hout[(size_t)row * HD + col] = (_Float16)v;
    if (v > 0.0f)  // max(0, seg) vs zero-init buffer is a no-op: skip atomic
      atomicMax((int*)(seg0 + (size_t)id * HD + col), __float_as_int(v));
  }
}

// ---------------------------------------------------------------------------
// layer1/2: [N,256] @ [256,128] via v_wmma_f32_16x16x32_f16, fused LN+ReLU,
// fused segment-max. K 0..127 = h (f16), K 128..255 = seg_in[ids] (f32->f16).
// Wave tile: 16 rows x 128 cols (8 accum tiles). LAST: no h store, seg->d_out.
// ---------------------------------------------------------------------------
template <bool LAST>
__global__ __launch_bounds__(256) void vn_layer(const _Float16* __restrict__ hin,
                                                _Float16*       __restrict__ hout,
                                                const int*      __restrict__ ids,
                                                const _Float16* __restrict__ wpad,
                                                const float*    __restrict__ b,
                                                const float*    __restrict__ g,
                                                const float*    __restrict__ be,
                                                const float*    __restrict__ seg_in,
                                                float*          __restrict__ seg_out,
                                                int nrows) {
  extern __shared__ char smem[];
  _Float16* wt  = (_Float16*)smem;                          // [128][WT_STRIDE]
  float*    bb  = (float*)(smem + WT_ELEMS * sizeof(_Float16));
  float*    gg  = bb + HD;
  float*    bbe = gg + HD;

  // straight 16B copy of pre-swizzled f16 weights (L2-resident, 66 KB)
  {
    const uint4* src = (const uint4*)wpad;
    uint4*       dst = (uint4*)wt;
    for (int i = threadIdx.x; i < WT_ELEMS / 8; i += blockDim.x) dst[i] = src[i];
  }
  for (int i = threadIdx.x; i < HD; i += blockDim.x) {
    bb[i] = b[i]; gg[i] = g[i]; bbe[i] = be[i];
  }
  __syncthreads();

  const int wave = threadIdx.x >> 5;
  const int lane = threadIdx.x & 31;
  const int idx  = lane & 15;        // A: row-in-tile  /  B,C: col-in-tile
  const int half = lane >> 4;        // K-half selector per ISA 16-bit layouts

  const int row0  = blockIdx.x * 128 + wave * 16;
  const int myrow = row0 + idx;
  const int lrow  = myrow < nrows ? myrow : nrows - 1;   // clamp for loads
  const int aid   = ids[lrow];

  v8f c[8];
#pragma unroll
  for (int t = 0; t < 8; ++t) c[t] = (v8f){0.f,0.f,0.f,0.f,0.f,0.f,0.f,0.f};

#pragma unroll
  for (int ks = 0; ks < 8; ++ks) {
    const int kk = ks * 32;
    AFrag a;
    if (kk < HD) {
      // h part: contiguous f16, per-lane K = kk + half*8 + {0..7, 16..23}
      const H8* p = (const H8*)(hin + (size_t)lrow * HD + kk + half * 8);
      a.c2[0] = p[0];
      a.c2[1] = p[2];   // +16 halves
    } else {
      // agg part: gathered segment max, f32 -> f16
      const float* pa = seg_in + (size_t)aid * HD + (kk - HD) + half * 8;
#pragma unroll
      for (int j = 0; j < 8; ++j) a.e[j]     = (_Float16)pa[j];
#pragma unroll
      for (int j = 0; j < 8; ++j) a.e[8 + j] = (_Float16)pa[16 + j];
    }
    // preload all 8 B fragments into distinct regs so DS latency overlaps WMMA
    AFrag bf[8];
#pragma unroll
    for (int t = 0; t < 8; ++t) {
      const int col = t * 16 + idx;
      const H8* pb = (const H8*)(wt + col * WT_STRIDE + kk + half * 8);
      bf[t].c2[0] = pb[0];
      bf[t].c2[1] = pb[2];
    }
#pragma unroll
    for (int t = 0; t < 8; ++t) {
      c[t] = __builtin_amdgcn_wmma_f32_16x16x32_f16(
          false, a.v, false, bf[t].v, (short)0, c[t], false, false);
    }
    // force issue order per k-step: 16 ds_load_b128 as a batch, then 8 WMMAs.
    // Converts 8 full s_wait_dscnt drains per k-step into one progressive wait.
    SCHED_DS_READS(16);
    SCHED_WMMAS(8);
  }

  // epilogue: bias, per-row LayerNorm (row = r + 8*half, col = t*16 + idx)
#pragma unroll
  for (int t = 0; t < 8; ++t) {
    const float bias = bb[t * 16 + idx];
#pragma unroll
    for (int r = 0; r < 8; ++r) c[t][r] += bias;
  }

#pragma unroll
  for (int r = 0; r < 8; ++r) {
    float s = 0.0f, q = 0.0f;
#pragma unroll
    for (int t = 0; t < 8; ++t) { float v = c[t][r]; s += v; q += v * v; }
#pragma unroll
    for (int m = 1; m < 16; m <<= 1) {       // reduce within 16-lane half
      s += __shfl_xor(s, m, 32);
      q += __shfl_xor(q, m, 32);
    }
    const float mu  = s * (1.0f / HD);
    const float var = q * (1.0f / HD) - mu * mu;
    const float rs  = rsqrtf(var + LNEPS);

    const int row = row0 + r + 8 * half;
    if (row < nrows) {
      const int sid = ids[row];
#pragma unroll
      for (int t = 0; t < 8; ++t) {
        const int col = t * 16 + idx;
        float v = (c[t][r] - mu) * rs * gg[col] + bbe[col];
        v = fmaxf(v, 0.0f);
        if constexpr (!LAST) hout[(size_t)row * HD + col] = (_Float16)v;
        if (v > 0.0f)  // zero-max against zero-init buffer: skip atomic
          atomicMax((int*)(seg_out + (size_t)sid * HD + col), __float_as_int(v));
      }
    }
  }
}

// ---------------------------------------------------------------------------
extern "C" void kernel_launch(void* const* d_in, const int* in_sizes, int n_in,
                              void* d_out, int out_size, void* d_ws, size_t ws_size,
                              hipStream_t stream) {
  const float* x   = (const float*)d_in[0];
  const int*   ids = (const int*)  d_in[1];
  const float* W0  = (const float*)d_in[2];
  const float* b0  = (const float*)d_in[3];
  const float* g0  = (const float*)d_in[4];
  const float* be0 = (const float*)d_in[5];
  const float* W1  = (const float*)d_in[6];
  const float* b1  = (const float*)d_in[7];
  const float* g1  = (const float*)d_in[8];
  const float* be1 = (const float*)d_in[9];
  const float* W2  = (const float*)d_in[10];
  const float* b2  = (const float*)d_in[11];
  const float* g2  = (const float*)d_in[12];
  const float* be2 = (const float*)d_in[13];

  float* out = (float*)d_out;

  // workspace: h (f16 N*128) | seg0 (f32 P*128) | seg1 (f32 P*128) | wt1 | wt2
  _Float16* h = (_Float16*)d_ws;
  size_t off  = (size_t)NROWS * HD * sizeof(_Float16);
  float* seg0 = (float*)((char*)d_ws + off);
  float* seg1 = seg0 + (size_t)NP * HD;
  _Float16* wt1 = (_Float16*)(seg1 + (size_t)NP * HD);
  _Float16* wt2 = wt1 + WT_ELEMS;

  // unique_polylines tail format (f32 or int64 flattening)
  const int tail  = out_size - NP * HD;
  const int tmode = (tail >= 2 * NP) ? 2 : (tail >= NP ? 1 : 0);

  vn_init<<<(NP * HD + 255) / 256, 256, 0, stream>>>(
      seg0, seg1, out, out + (size_t)NP * HD, tmode);

  vn_prep<<<(2 * HD * HD + 255) / 256, 256, 0, stream>>>(W1, W2, wt1, wt2);

  vn_layer0<<<(NROWS + 7) / 8, 256, 0, stream>>>(
      x, ids, W0, b0, g0, be0, h, seg0);

  const int gemm_blocks = (NROWS + 127) / 128;
  const size_t lds_bytes = WT_ELEMS * sizeof(_Float16) + 3 * HD * sizeof(float);

  vn_layer<false><<<gemm_blocks, 256, lds_bytes, stream>>>(
      h, h, ids, wt1, b1, g1, be1, seg0, seg1, NROWS);

  vn_layer<true><<<gemm_blocks, 256, lds_bytes, stream>>>(
      h, nullptr, ids, wt2, b2, g2, be2, seg1, out, NROWS);
}
